// TALayer_1984274891484
// MI455X (gfx1250) — matrix operand order repaired and checked by reading, not compiled, not run on hardware
//
#include <hip/hip_runtime.h>

typedef __attribute__((ext_vector_type(16))) __bf16 v16bf;
typedef __attribute__((ext_vector_type(8)))  float  v8f;

#define IN_C   64
#define OUT_C  64
#define KTAPS  9
#define TILE_T 256                 // t-values per workgroup (8 waves x 2 x 16)
#define XROWS  320                 // staged t rows = TILE_T + 64 halo margin
#define XPITCH 72                  // bf16 per staged row (64 ch + 8 pad)
#define WELEMS (KTAPS * 2 * 4 * 32 * 16)   // 36864 = 64*64*9

// Load a 32-byte per-lane fragment (16 bf16) from LDS as two 16B chunks.
__device__ __forceinline__ v16bf ld_frag32(const __bf16* p) {
  union { uint4 u[2]; v16bf v; } un;
  const uint4* q = (const uint4*)p;
  un.u[0] = q[0];
  un.u[1] = q[1];
  return un.v;
}

__global__ __launch_bounds__(256)
void tada_conv_bf16_wmma(const float* __restrict__ x,
                         const float* __restrict__ w,
                         const float* __restrict__ bias,
                         float* __restrict__ out, int T) {
  __shared__ __bf16 lds_w[WELEMS];            // A fragments, 73728 B
  __shared__ __bf16 lds_x[XROWS * XPITCH];    // transposed x tile, 46080 B
  __shared__ float  lds_b[OUT_C];

  const int tid  = threadIdx.x;
  const int bIdx = blockIdx.y;
  const int wgT0 = blockIdx.x * TILE_T;

  // ---- Stage weights into 16x32 bf16 A fragments, contiguous k order (same as B) ----
  // frag f = ((j*2 + ch)*4 + mb); per lane l: 16 bf16 elements e.
  // lanes 0-15: K = e ; lanes 16-31: K = e+16
  for (int idx = tid; idx < WELEMS; idx += 256) {
    int e  = idx & 15;
    int l  = (idx >> 4) & 31;
    int f  = idx >> 9;
    int j  = f >> 3;
    int ch = (f >> 2) & 1;
    int mb = f & 3;
    int hi = l >> 4;
    int m  = mb * 16 + (l & 15);
    int kl = e + hi * 16;
    int c  = ch * 32 + kl;
    lds_w[idx] = (__bf16)w[(m * IN_C + c) * KTAPS + j];
  }

  // ---- Stage x tile transposed: lds_x[row][c], row = t - (wgT0-32) ----
  const float* xb = x + (size_t)bIdx * IN_C * T;
  for (int e = tid; e < IN_C * XROWS; e += 256) {
    int c = e / XROWS;
    int r = e - c * XROWS;
    int t = wgT0 - 32 + r;
    float v = 0.0f;
    if (t >= 0 && t < T) v = xb[(size_t)c * T + t];
    lds_x[r * XPITCH + c] = (__bf16)v;
  }

  if (tid < OUT_C) lds_b[tid] = bias[tid];
  __syncthreads();

  // ---- Per-wave 64(M) x 32(N=t) output: two 16-t tiles sharing A frags ----
  const int wave = tid >> 5;
  const int lane = tid & 31;
  const int n    = lane & 15;
  const int hi   = lane >> 4;
  const int t0   = wgT0 + wave * 32;

  v8f acc0[4] = {};
  v8f acc1[4] = {};
  const int dtap[KTAPS] = {-17, -16, -15, -1, 0, 1, 15, 16, 17};

  #pragma unroll
  for (int j = 0; j < KTAPS; ++j) {
    const int row0 = wave * 32 + 32 + n + dtap[j];   // in [15, 304]
    #pragma unroll
    for (int ch = 0; ch < 2; ++ch) {
      // B fragments 32x16: lane gives N=t; 16 contiguous channels per lane.
      const int cOff = ch * 32 + hi * 16;
      v16bf bfrag0 = ld_frag32(&lds_x[row0 * XPITCH + cOff]);
      v16bf bfrag1 = ld_frag32(&lds_x[(row0 + 16) * XPITCH + cOff]);
      #pragma unroll
      for (int mb = 0; mb < 4; ++mb) {
        const int f = (j * 2 + ch) * 4 + mb;
        v16bf afrag = ld_frag32(&lds_w[f * 512 + lane * 16]);
        acc0[mb] = __builtin_amdgcn_wmma_f32_16x16x32_bf16(
            false, afrag, false, bfrag0, (short)0, acc0[mb], false, false);
        acc1[mb] = __builtin_amdgcn_wmma_f32_16x16x32_bf16(
            false, afrag, false, bfrag1, (short)0, acc1[mb], false, false);
      }
    }
  }

  // ---- Store: C/D layout VGPR v -> M = v (lanes 0-15) / v+8 (lanes 16-31)
  float* ob = out + (size_t)bIdx * OUT_C * T;
  #pragma unroll
  for (int mb = 0; mb < 4; ++mb) {
    #pragma unroll
    for (int v = 0; v < 8; ++v) {
      int o = mb * 16 + v + hi * 8;
      float bv = lds_b[o];
      ob[(size_t)o * T + (t0 + n)]      = acc0[mb][v] + bv;
      ob[(size_t)o * T + (t0 + 16 + n)] = acc1[mb][v] + bv;
    }
  }
}

extern "C" void kernel_launch(void* const* d_in, const int* in_sizes, int n_in,
                              void* d_out, int out_size, void* d_ws, size_t ws_size,
                              hipStream_t stream) {
  const float* x    = (const float*)d_in[0];
  const float* w    = (const float*)d_in[1];
  const float* bias = (const float*)d_in[2];
  float* out = (float*)d_out;

  const int B = 8;
  const int T = in_sizes[0] / (B * IN_C);   // 65536

  dim3 grid(T / TILE_T, B);
  tada_conv_bf16_wmma<<<grid, 256, 0, stream>>>(x, w, bias, out, T);
}
